// MultiTaskContrastiveLoss_76141180223511
// MI455X (gfx1250) — compile-verified, matrix operand order checked by reading, not run on hardware
//
#include <hip/hip_runtime.h>
#include <cstdint>
#include <cstddef>

// ---------------------------------------------------------------------------
// MultiTaskContrastiveLoss for MI455X (gfx1250, wave32, WMMA 16x16x32 bf16,
// async global->LDS staging on the ASYNCcnt path).
// ---------------------------------------------------------------------------

namespace {

constexpr int   kB      = 8192;
constexpr int   kHid    = 768;
constexpr float kTau    = 0.07f;
constexpr float kMargin = 0.2f;
constexpr float kEps    = 1e-6f;

typedef __attribute__((ext_vector_type(8)))  __bf16 v8bf;
typedef __attribute__((ext_vector_type(16))) __bf16 v16bf;
typedef __attribute__((ext_vector_type(8)))  float  v8f;

__device__ inline v8f wmma_bf16(const v16bf& a, const v16bf& b, const v8f& c) {
  return __builtin_amdgcn_wmma_f32_16x16x32_bf16(
      false, a, false, b, (short)0, c, false, false);
}

// Async DMA of one 16-byte chunk per lane: global (saddr + 32-bit voffset,
// GVS mode) -> LDS byte address held in a VGPR.  Completion tracked by
// ASYNCcnt (CDNA5 ISA).
__device__ inline void async_copy16(uint32_t lds_addr, uint32_t goff,
                                    uint64_t gbase) {
  asm volatile("global_load_async_to_lds_b128 %0, %1, %2"
               :: "v"(lds_addr), "v"(goff), "s"(gbase)
               : "memory");
}

// Low 32 bits of a generic pointer to __shared__ data are the wave-relative
// LDS byte address (ISA §10.2: LDS aperture uses addr[31:0]).
__device__ inline uint32_t lds_addr_of(const void* p) {
  return (uint32_t)(size_t)p;
}

// Load a 16x32 bf16 fragment (A-matrix layout, ISA 7.12.2) straight from a
// row-major [rows, ld] bf16 matrix at (row0, k0): lanes 0-15 hold row M=lane
// with K 0..7 / 16..23; lanes 16-31 hold K 8..15 / 24..31.  Two contiguous
// 16-byte loads per lane.  Also yields the B-fragment of B = X^T.
__device__ inline v16bf load_frag16x32(const __bf16* __restrict__ p, int ld,
                                       int row0, int k0) {
  const int lane = threadIdx.x & 31;
  const int half = lane >> 4;
  const int r    = lane & 15;
  const __bf16* base = p + (size_t)(row0 + r) * (size_t)ld + (k0 + half * 8);
  v8bf lo = *reinterpret_cast<const v8bf*>(base);
  v8bf hi = *reinterpret_cast<const v8bf*>(base + 16);
  v16bf out;
#pragma unroll
  for (int i = 0; i < 8; ++i) { out[i] = lo[i]; out[i + 8] = hi[i]; }
  return out;
}

// Same fragment, but from an LDS tile whose rows are 80B apart (64B of k-data
// + 16B pad -> conflict-free ds_load_b128 across the 16 row-lanes).
constexpr int kLdsRow = 80;

__device__ inline v16bf lds_frag16x32(const char* base, int rowInTile) {
  const int lane = threadIdx.x & 31;
  const int half = lane >> 4;
  const int r    = lane & 15;
  const char* p = base + (rowInTile + r) * kLdsRow + half * 16;
  v8bf lo = *reinterpret_cast<const v8bf*>(p);        // K half*8 .. +7
  v8bf hi = *reinterpret_cast<const v8bf*>(p + 32);   // K 16+half*8 .. +23
  v16bf out;
#pragma unroll
  for (int i = 0; i < 8; ++i) { out[i] = lo[i]; out[i + 8] = hi[i]; }
  return out;
}

// ---------------------------------------------------------------------------
// Elementwise / setup kernels
// ---------------------------------------------------------------------------

__global__ void zero_f32_kernel(float* __restrict__ p, int n) {
  int i = blockIdx.x * blockDim.x + threadIdx.x;
  if (i < n) p[i] = 0.0f;
}

__global__ void f32_to_bf16_kernel(const float* __restrict__ src,
                                   __bf16* __restrict__ dst, int n) {
  int i = blockIdx.x * blockDim.x + threadIdx.x;
  if (i < n) dst[i] = (__bf16)src[i];
}

// W fp32 [K,N] row-major -> Wt bf16 [N,K] row-major (coalesced writes).
__global__ void transpose_to_bf16_kernel(const float* __restrict__ src,
                                         __bf16* __restrict__ dst, int K, int N) {
  int i = blockIdx.x * blockDim.x + threadIdx.x;
  if (i >= K * N) return;
  int n = i / K;
  int k = i - n * K;
  dst[i] = (__bf16)src[(size_t)k * N + n];
}

// ---------------------------------------------------------------------------
// WMMA GEMM with double-buffered async LDS staging.
//   C[M,N] = act(A[M,K] @ Bt[N,K]^T + bias)          (bf16 in, bf16 out)
// Block = 256 threads = 8 waves; block tile 256x64; wave tile 32x64 (2x4 of
// 16x16).  Per K-step of 32: stage A 256x32 + B 64x32 (5 async b128 per wave),
// prefetch the next slab into the other buffer, wait asynccnt<=5, barrier,
// ds_load fragments, 8 wmma per wave.  K must be a multiple of 32; M of 256;
// N of 64.
// ---------------------------------------------------------------------------

__device__ inline void store_tile_bf16(__bf16* __restrict__ C, int N,
                                       int m0, int n0, const v8f& acc,
                                       const float* __restrict__ bias, int relu) {
  const int lane = threadIdx.x & 31;
  const int half = lane >> 4;
  const int c    = lane & 15;
  const int col  = n0 + c;
  const float bi = bias[col];
#pragma unroll
  for (int v = 0; v < 8; ++v) {
    // C/D layout: VGPR v, lanes 0-15 -> row m0+v; lanes 16-31 -> row m0+8+v.
    const int row = m0 + v + 8 * half;
    float x = acc[v] + bi;
    if (relu) x = fmaxf(x, 0.0f);
    C[(size_t)row * N + col] = (__bf16)x;
  }
}

__global__ __launch_bounds__(256) void gemm_bf16_kernel(
    const __bf16* __restrict__ A, const __bf16* __restrict__ Bt,
    const float* __restrict__ bias, __bf16* __restrict__ C,
    int M, int K, int N, int relu) {
  constexpr int kABytes   = 256 * kLdsRow;          // 20480
  constexpr int kBBytes   = 64 * kLdsRow;           //  5120
  constexpr int kBufBytes = kABytes + kBBytes;      // 25600
  __shared__ __align__(16) char smem[2 * kBufBytes];
  (void)M;

  const int tid  = threadIdx.x;
  const int wave = tid >> 5;
  const int lane = tid & 31;
  const int Mblk = blockIdx.x * 256;
  const int Nblk = blockIdx.y * 64;

  const uint64_t aBase    = (uint64_t)(size_t)A;
  const uint64_t bBase    = (uint64_t)(size_t)Bt;
  const uint32_t smemBase = lds_addr_of(smem);

  // Stage one K-slab (k0..k0+31) of the block tile into LDS buffer `buf`.
  // 5 async b128 ops per wave: 4 for A (1024 chunks / 256 lanes), 1 for B.
  auto stage = [&](int buf, int k0) {
    const uint32_t aOff = smemBase + buf * kBufBytes;
    const uint32_t bOff = aOff + kABytes;
#pragma unroll
    for (int i = 0; i < 4; ++i) {
      const int q   = (wave * 4 + i) * 32 + lane;     // [0,1024)
      const int row = q >> 2, seg = q & 3;            // 4x16B chunks per row
      const uint32_t g = (uint32_t)(((Mblk + row) * K + k0 + seg * 8) * 2);
      async_copy16(aOff + row * kLdsRow + seg * 16, g, aBase);
    }
    {
      const int q   = wave * 32 + lane;               // [0,256)
      const int row = q >> 2, seg = q & 3;
      const uint32_t g = (uint32_t)(((Nblk + row) * K + k0 + seg * 8) * 2);
      async_copy16(bOff + row * kLdsRow + seg * 16, g, bBase);
    }
  };

  v8f acc[2][4] = {};
  const int m0w = wave * 32;
  const int nk  = K >> 5;

  stage(0, 0);
  for (int i = 0; i < nk; ++i) {
    const int cur = i & 1;
    if (i + 1 < nk) {
      stage(cur ^ 1, (i + 1) * 32);                   // prefetch next slab
      // Async ops of one wave complete in order: <=5 outstanding means the
      // current slab's 5 are done (only the prefetch remains in flight).
      asm volatile("s_wait_asynccnt 0x5" ::: "memory");
    } else {
      asm volatile("s_wait_asynccnt 0x0" ::: "memory");
    }
    __syncthreads();                                  // publish staged slab

    const char* aT = smem + cur * kBufBytes;
    const char* bT = smem + cur * kBufBytes + kABytes;
    v16bf aF0 = lds_frag16x32(aT, m0w);
    v16bf aF1 = lds_frag16x32(aT, m0w + 16);
    v16bf bF0 = lds_frag16x32(bT, 0);
    v16bf bF1 = lds_frag16x32(bT, 16);
    v16bf bF2 = lds_frag16x32(bT, 32);
    v16bf bF3 = lds_frag16x32(bT, 48);
    acc[0][0] = wmma_bf16(aF0, bF0, acc[0][0]);
    acc[0][1] = wmma_bf16(aF0, bF1, acc[0][1]);
    acc[0][2] = wmma_bf16(aF0, bF2, acc[0][2]);
    acc[0][3] = wmma_bf16(aF0, bF3, acc[0][3]);
    acc[1][0] = wmma_bf16(aF1, bF0, acc[1][0]);
    acc[1][1] = wmma_bf16(aF1, bF1, acc[1][1]);
    acc[1][2] = wmma_bf16(aF1, bF2, acc[1][2]);
    acc[1][3] = wmma_bf16(aF1, bF3, acc[1][3]);

    __syncthreads();   // all reads of this buffer done before it is re-staged
  }

#pragma unroll
  for (int sub = 0; sub < 2; ++sub)
#pragma unroll
    for (int j = 0; j < 4; ++j)
      store_tile_bf16(C, N, Mblk + m0w + sub * 16, Nblk + j * 16,
                      acc[sub][j], bias, relu);
}

// ---------------------------------------------------------------------------
// L2-normalize 128-wide rows (fp32 math) and scatter into dst at colOff.
// One wave per row; 4 elements per lane; full-wave shuffle reduction.
// ---------------------------------------------------------------------------

__global__ __launch_bounds__(256) void l2norm_scatter_kernel(
    const __bf16* __restrict__ src, __bf16* __restrict__ dst,
    int ldDst, int colOff) {
  const int wave = threadIdx.x >> 5;
  const int lane = threadIdx.x & 31;
  const int row  = blockIdx.x * 8 + wave;
  const __bf16* s = src + (size_t)row * 128 + lane * 4;
  float x0 = (float)s[0], x1 = (float)s[1], x2 = (float)s[2], x3 = (float)s[3];
  float ss = x0 * x0 + x1 * x1 + x2 * x2 + x3 * x3;
#pragma unroll
  for (int m = 1; m < 32; m <<= 1) ss += __shfl_xor(ss, m, 32);
  const float inv = 1.0f / sqrtf(ss);
  __bf16* d = dst + (size_t)row * ldDst + colOff + lane * 4;
  d[0] = (__bf16)(x0 * inv);
  d[1] = (__bf16)(x1 * inv);
  d[2] = (__bf16)(x2 * inv);
  d[3] = (__bf16)(x3 * inv);
}

// ---------------------------------------------------------------------------
// Similarity pass: sim = (z z^T)/tau fused with exp + masked row reductions.
// Wave w of block bx owns row tile m0 = (bx*8+w)*16 and keeps its 4 A-frags
// of z resident (z is 2MB -> column-tile B-frags stream from L2).  grid.y
// splits the 512 column tiles into `colGroups` slices.
// ---------------------------------------------------------------------------

__global__ __launch_bounds__(256) void sim_accum_kernel(
    const __bf16* __restrict__ Z, const int* __restrict__ labels,
    float* __restrict__ sum_exp, float* __restrict__ pos_sum, int colGroups) {
  const int wave = threadIdx.x >> 5;
  const int lane = threadIdx.x & 31;
  const int half = lane >> 4;
  const int c    = lane & 15;
  const int m0   = (blockIdx.x * 8 + wave) * 16;

  const v16bf a0 = load_frag16x32(Z, 128, m0, 0);
  const v16bf a1 = load_frag16x32(Z, 128, m0, 32);
  const v16bf a2 = load_frag16x32(Z, 128, m0, 64);
  const v16bf a3 = load_frag16x32(Z, 128, m0, 96);

  int labM[8];
#pragma unroll
  for (int v = 0; v < 8; ++v) labM[v] = labels[m0 + v + 8 * half];

  float sacc[8] = {0, 0, 0, 0, 0, 0, 0, 0};
  float pacc[8] = {0, 0, 0, 0, 0, 0, 0, 0};

  const int tilesPerGroup = (kB / 16) / colGroups;
  const int t0 = blockIdx.y * tilesPerGroup;
  for (int t = 0; t < tilesPerGroup; ++t) {
    const int n0 = (t0 + t) * 16;
    // B = z^T, so the B-fragment is the A-layout fragment of z's rows n0..+15.
    const v16bf b0 = load_frag16x32(Z, 128, n0, 0);
    const v16bf b1 = load_frag16x32(Z, 128, n0, 32);
    const v16bf b2 = load_frag16x32(Z, 128, n0, 64);
    const v16bf b3 = load_frag16x32(Z, 128, n0, 96);
    v8f acc = {};
    acc = wmma_bf16(a0, b0, acc);
    acc = wmma_bf16(a1, b1, acc);
    acc = wmma_bf16(a2, b2, acc);
    acc = wmma_bf16(a3, b3, acc);

    const int col  = n0 + c;
    const int labN = labels[col];
#pragma unroll
    for (int v = 0; v < 8; ++v) {
      const int row = m0 + v + 8 * half;
      const float e = __expf(acc[v] * (1.0f / kTau));
      sacc[v] += e;                                        // includes diagonal
      if (labM[v] == labN && row != col) pacc[v] += e;     // positives, no diag
    }
  }

  // Lanes {0..15} and {16..31} each cover one set of 8 rows; reduce the 16
  // column-lanes of each half together (xor masks < 16 stay within the half).
#pragma unroll
  for (int v = 0; v < 8; ++v) {
#pragma unroll
    for (int m = 1; m < 16; m <<= 1) {
      sacc[v] += __shfl_xor(sacc[v], m, 32);
      pacc[v] += __shfl_xor(pacc[v], m, 32);
    }
  }
  if (c == 0) {
#pragma unroll
    for (int v = 0; v < 8; ++v) {
      const int row = m0 + v + 8 * half;
      atomicAdd(&sum_exp[row], sacc[v]);
      atomicAdd(&pos_sum[row], pacc[v]);
    }
  }
}

// stats[0]=scl_sum, stats[1]=n_valid, stats[2]=tri_sum
__global__ void row_loss_kernel(const float* __restrict__ sum_exp,
                                const float* __restrict__ pos_sum,
                                float* __restrict__ stats) {
  int i = blockIdx.x * blockDim.x + threadIdx.x;
  if (i >= kB) return;
  const float ps = pos_sum[i];
  if (ps > 0.0f) {
    atomicAdd(&stats[0], __logf(sum_exp[i]) - __logf(ps));
    atomicAdd(&stats[1], 1.0f);
  }
}

// Triplet loss: wave per row over 768 fp32 elements.
__global__ __launch_bounds__(256) void triplet_kernel(
    const float* __restrict__ A, const float* __restrict__ P,
    const float* __restrict__ Ng, float* __restrict__ stats) {
  const int wave = threadIdx.x >> 5;
  const int lane = threadIdx.x & 31;
  const int row  = blockIdx.x * 8 + wave;
  const float* a = A  + (size_t)row * kHid;
  const float* p = P  + (size_t)row * kHid;
  const float* n = Ng + (size_t)row * kHid;
  float sp = 0.0f, sn = 0.0f;
  for (int j = lane; j < kHid; j += 32) {
    const float av = a[j];
    const float dp = av - p[j] + kEps;
    const float dn = av - n[j] + kEps;
    sp += dp * dp;
    sn += dn * dn;
  }
#pragma unroll
  for (int m = 1; m < 32; m <<= 1) {
    sp += __shfl_xor(sp, m, 32);
    sn += __shfl_xor(sn, m, 32);
  }
  if (lane == 0) {
    const float t = sqrtf(sp) - sqrtf(sn) + kMargin;
    if (t > 0.0f) atomicAdd(&stats[2], t);
  }
}

__global__ void finalize_kernel(const float* __restrict__ stats,
                                float* __restrict__ out) {
  const float nv = fmaxf(stats[1], 1.0f);
  out[0] = stats[0] / nv + 0.5f * (stats[2] / (float)kB);
}

}  // namespace

// ---------------------------------------------------------------------------
// Host launcher
// ---------------------------------------------------------------------------

extern "C" void kernel_launch(void* const* d_in, const int* in_sizes, int n_in,
                              void* d_out, int out_size, void* d_ws, size_t ws_size,
                              hipStream_t stream) {
  (void)in_sizes; (void)n_in; (void)out_size; (void)ws_size;

  const float* branch_in[3] = {(const float*)d_in[0], (const float*)d_in[1],
                               (const float*)d_in[2]};
  const float* anchor   = (const float*)d_in[3];
  const float* positive = (const float*)d_in[4];
  const float* negative = (const float*)d_in[5];
  const int*   labels   = (const int*)d_in[6];
  // aspect_params @7..12, opinion @13..18, sentiment @19..24, combined @25..28
  const float* Wc1 = (const float*)d_in[25];
  const float* bc1 = (const float*)d_in[26];
  const float* Wc2 = (const float*)d_in[27];
  const float* bc2 = (const float*)d_in[28];

  // ---- workspace carve-out (256B aligned) ----
  char* ws = (char*)d_ws;
  size_t off = 0;
  auto alloc = [&](size_t bytes) -> void* {
    void* p = ws + off;
    off += (bytes + 255) & ~(size_t)255;
    return p;
  };

  float*  sum_exp = (float*)alloc(kB * sizeof(float));
  float*  pos_sum = (float*)alloc(kB * sizeof(float));
  float*  stats   = (float*)alloc(4 * sizeof(float));

  // Transposed bf16 weights.  Per-branch dims: 768->384->192->128.
  __bf16* Wt1[3]; __bf16* Wt2[3]; __bf16* Wt3[3];
  for (int b = 0; b < 3; ++b) {
    Wt1[b] = (__bf16*)alloc((size_t)384 * 768 * 2);
    Wt2[b] = (__bf16*)alloc((size_t)192 * 384 * 2);
    Wt3[b] = (__bf16*)alloc((size_t)128 * 192 * 2);
  }
  __bf16* WtC1 = (__bf16*)alloc((size_t)256 * 384 * 2);
  __bf16* WtC2 = (__bf16*)alloc((size_t)128 * 256 * 2);

  __bf16* Xbf  = (__bf16*)alloc((size_t)kB * 768 * 2);
  __bf16* H1   = (__bf16*)alloc((size_t)kB * 384 * 2);  // reused as C1 [kB,256]
  __bf16* H2   = (__bf16*)alloc((size_t)kB * 192 * 2);  // reused as Zp [kB,128]
  __bf16* E    = (__bf16*)alloc((size_t)kB * 128 * 2);  // reused as Z  [kB,128]
  __bf16* Comb = (__bf16*)alloc((size_t)kB * 384 * 2);

  __bf16* C1 = H1;
  __bf16* Zp = H2;
  __bf16* Z  = E;

  // ---- zero accumulators (atomics accumulate across graph replays) ----
  zero_f32_kernel<<<(kB + 255) / 256, 256, 0, stream>>>(sum_exp, kB);
  zero_f32_kernel<<<(kB + 255) / 256, 256, 0, stream>>>(pos_sum, kB);
  zero_f32_kernel<<<1, 256, 0, stream>>>(stats, 4);

  // ---- weight transpose+convert ----
  auto xpose = [&](const float* src, __bf16* dst, int K, int N) {
    int n = K * N;
    transpose_to_bf16_kernel<<<(n + 255) / 256, 256, 0, stream>>>(src, dst, K, N);
  };
  for (int b = 0; b < 3; ++b) {
    const int base = 7 + 6 * b;
    xpose((const float*)d_in[base + 0], Wt1[b], 768, 384);
    xpose((const float*)d_in[base + 2], Wt2[b], 384, 192);
    xpose((const float*)d_in[base + 4], Wt3[b], 192, 128);
  }
  xpose(Wc1, WtC1, 384, 256);
  xpose(Wc2, WtC2, 256, 128);

  auto gemm = [&](const __bf16* A, const __bf16* Bt, const float* bias,
                  __bf16* C, int M, int K, int N, int relu) {
    dim3 grid(M / 256, N / 64);
    gemm_bf16_kernel<<<grid, 256, 0, stream>>>(A, Bt, bias, C, M, K, N, relu);
  };

  // ---- three branch MLPs + normalize-scatter into Comb [kB, 384] ----
  for (int b = 0; b < 3; ++b) {
    const int base = 7 + 6 * b;
    const float* b1 = (const float*)d_in[base + 1];
    const float* b2 = (const float*)d_in[base + 3];
    const float* b3 = (const float*)d_in[base + 5];
    f32_to_bf16_kernel<<<(kB * 768 + 255) / 256, 256, 0, stream>>>(
        branch_in[b], Xbf, kB * 768);
    gemm(Xbf, Wt1[b], b1, H1, kB, 768, 384, 1);
    gemm(H1,  Wt2[b], b2, H2, kB, 384, 192, 1);
    gemm(H2,  Wt3[b], b3, E,  kB, 192, 128, 0);
    l2norm_scatter_kernel<<<kB / 8, 256, 0, stream>>>(E, Comb, 384, 128 * b);
  }

  // ---- combined MLP + normalize -> z ----
  gemm(Comb, WtC1, bc1, C1, kB, 384, 256, 1);
  gemm(C1,   WtC2, bc2, Zp, kB, 256, 128, 0);
  l2norm_scatter_kernel<<<kB / 8, 256, 0, stream>>>(Zp, Z, 128, 0);

  // ---- SCL: fused Gram/exp/row-reduction, then per-row loss ----
  const int colGroups = 8;
  dim3 simGrid(kB / (16 * 8), colGroups);
  sim_accum_kernel<<<simGrid, 256, 0, stream>>>(Z, labels, sum_exp, pos_sum,
                                                colGroups);
  row_loss_kernel<<<(kB + 255) / 256, 256, 0, stream>>>(sum_exp, pos_sum, stats);

  // ---- triplet + finalize ----
  triplet_kernel<<<kB / 8, 256, 0, stream>>>(anchor, positive, negative, stats);
  finalize_kernel<<<1, 1, 0, stream>>>(stats, (float*)d_out);
}
